// Attention_11802570129977
// MI455X (gfx1250) — matrix operand order repaired and checked by reading, not compiled, not run on hardware
//
#include <hip/hip_runtime.h>
#include <math.h>

#define S_LEN 4096
#define DIM 256
#define NB 32
#define LDS_STRIDE 258               // padded ushort stride (bank-conflict-friendly)
#define ENC_ROWS 256                 // rows per workgroup (32 per wave)

typedef __attribute__((ext_vector_type(16))) __bf16 v16bf;
typedef __attribute__((ext_vector_type(8)))  float  v8f;

union Frag16 { unsigned int u[8]; v16bf v; };

__device__ __forceinline__ unsigned short f2bf(float f) {
    unsigned int u = __builtin_bit_cast(unsigned int, f);
    unsigned int r = (u + 0x7FFFu + ((u >> 16) & 1u)) >> 16;   // RNE
    return (unsigned short)r;
}

__device__ __forceinline__ unsigned int pack_bf16(float x, float y) {
#if __has_builtin(__builtin_amdgcn_cvt_pk_bf16_f32)
    auto p = __builtin_amdgcn_cvt_pk_bf16_f32(x, y);
    return __builtin_bit_cast(unsigned int, p);
#else
    return (unsigned int)f2bf(x) | ((unsigned int)f2bf(y) << 16);
#endif
}

__device__ __forceinline__ float tanh_fast(float x) {
#if __has_builtin(__builtin_amdgcn_tanhf)
    return __builtin_amdgcn_tanhf(x);
#elif __has_builtin(__builtin_amdgcn_tanh_f32)
    return __builtin_amdgcn_tanh_f32(x);
#else
    // tanh(x) = 1 - 2/(e^{2x}+1); branch-free, v_exp_f32 + v_rcp_f32
    float t = __builtin_amdgcn_exp2f(x * 2.8853900817779268f);
    return 1.0f - 2.0f * __builtin_amdgcn_rcpf(t + 1.0f);
#endif
}

// ---------------- K1: qb[b,a] = dec[b,:]@W1[:,a] + b1[a] + b2[a] ----------------
__global__ __launch_bounds__(256) void qb_kernel(
    const float* __restrict__ dec, const float* __restrict__ w1,
    const float* __restrict__ b1,  const float* __restrict__ b2,
    float* __restrict__ qb)
{
    int b = blockIdx.x, a = threadIdx.x;
    float acc = b1[a] + b2[a];
    const float* dp = dec + b * DIM;
    #pragma unroll 8
    for (int e = 0; e < DIM; ++e) acc = fmaf(dp[e], w1[e * DIM + a], acc);
    qb[b * DIM + a] = acc;
}

// ---------------- K1b: W2 -> bf16, transposed to [a][e] ----------------
__global__ __launch_bounds__(256) void w2t_kernel(
    const float* __restrict__ w2, unsigned short* __restrict__ w2bf)
{
    int i = blockIdx.x * 256 + threadIdx.x;   // i = a*256 + e
    int a = i >> 8, e = i & 255;
    w2bf[i] = f2bf(w2[e * DIM + a]);
}

// ---------------- K2: score[b,s] = v . tanh(qb + enc@W2) + v_bias ----------------
__global__ __launch_bounds__(256) void score_kernel(
    const float* __restrict__ enc, const unsigned short* __restrict__ w2bf,
    const float* __restrict__ qb,  const float* __restrict__ vker,
    const float* __restrict__ vbp, float* __restrict__ scores)
{
    extern __shared__ unsigned char ldsraw[];
    unsigned short* encT = (unsigned short*)ldsraw;     // [256][LDS_STRIDE]
    unsigned short* w2T  = encT + ENC_ROWS * LDS_STRIDE;// [256][LDS_STRIDE], [a][e]
    float* qvL = (float*)(w2T + DIM * LDS_STRIDE);      // [256]
    float* vvL = qvL + DIM;                             // [256]

    const int b  = blockIdx.y;
    const int s0 = blockIdx.x * ENC_ROWS;
    const int tid = threadIdx.x;

    // stage encoder tile (f32 -> bf16), coalesced float4 loads
    const float* encBase = enc + ((size_t)(b * S_LEN + s0)) * DIM;
    for (int i = tid; i < ENC_ROWS * (DIM / 4); i += 256) {
        int r = i >> 6, cq = (i & 63) << 2;
        float4 f = *(const float4*)(encBase + r * DIM + cq);
        unsigned int* p = (unsigned int*)(encT + r * LDS_STRIDE + cq);
        p[0] = pack_bf16(f.x, f.y);
        p[1] = pack_bf16(f.z, f.w);
    }
    // stage W2^T bf16 (pre-converted), uint4 loads
    for (int i = tid; i < DIM * (DIM / 8); i += 256) {
        int r = i >> 5, c8 = (i & 31) << 3;
        uint4 d = *(const uint4*)(w2bf + r * DIM + c8);
        unsigned int* p = (unsigned int*)(w2T + r * LDS_STRIDE + c8);
        p[0] = d.x; p[1] = d.y; p[2] = d.z; p[3] = d.w;
    }
    // stage qb row and v
    qvL[tid] = qb[b * DIM + tid];
    vvL[tid] = vker[tid];
    __syncthreads();

    const int wv = tid >> 5;          // wave id 0..7
    const int l  = tid & 31;          // lane
    const int lr = l & 15;            // N / row sub-index
    const int lh = l >> 4;            // lane half
    const int r0 = wv * 32;           // this wave's 32 rows (two 16-row slices)

    // A fragments: full K=256 for two 16-row slices
    Frag16 A0[8], A1[8];
    #pragma unroll
    for (int k = 0; k < 8; ++k) {
        const unsigned short* p0 = encT + (r0 + lr) * LDS_STRIDE      + k * 32 + lh * 8;
        const unsigned short* p1 = encT + (r0 + 16 + lr) * LDS_STRIDE + k * 32 + lh * 8;
        #pragma unroll
        for (int v = 0; v < 4; ++v) {
            A0[k].u[v]     = *(const unsigned int*)(p0 + 2 * v);
            A0[k].u[4 + v] = *(const unsigned int*)(p0 + 16 + 2 * v);
            A1[k].u[v]     = *(const unsigned int*)(p1 + 2 * v);
            A1[k].u[4 + v] = *(const unsigned int*)(p1 + 16 + 2 * v);
        }
    }

    const float vbias = vbp[0];
    float sc0[8] = {0.f,0.f,0.f,0.f,0.f,0.f,0.f,0.f};
    float sc1[8] = {0.f,0.f,0.f,0.f,0.f,0.f,0.f,0.f};

    for (int n = 0; n < 16; ++n) {            // 16 column tiles of the att dim
        v8f C0 = {}, C1 = {};
        const unsigned short* colp = w2T + (n * 16 + lr) * LDS_STRIDE + lh * 8;
        #pragma unroll
        for (int k = 0; k < 8; ++k) {         // K = 256 in 8 steps
            Frag16 Bf;
            #pragma unroll
            for (int v = 0; v < 8; ++v)
                Bf.u[v] = *(const unsigned int*)(colp + k * 32 + 2 * v + (v >= 4 ? 8 : 0));
            C0 = __builtin_amdgcn_wmma_f32_16x16x32_bf16(
                    false, A0[k].v, false, Bf.v, (short)0, C0, false, false);
            C1 = __builtin_amdgcn_wmma_f32_16x16x32_bf16(
                    false, A1[k].v, false, Bf.v, (short)0, C1, false, false);
        }
        const int a = n * 16 + lr;
        const float qv = qvL[a];
        const float vv = vvL[a];
        #pragma unroll
        for (int j = 0; j < 8; ++j) sc0[j] += vv * tanh_fast(C0[j] + qv);
        #pragma unroll
        for (int j = 0; j < 8; ++j) sc1[j] += vv * tanh_fast(C1[j] + qv);
    }

    // reduce across the 16 N-lanes of each half (C layout: M = j + 8*lh, N = lr)
    #pragma unroll
    for (int j = 0; j < 8; ++j) {
        float s = sc0[j];
        s += __shfl_xor(s, 1); s += __shfl_xor(s, 2);
        s += __shfl_xor(s, 4); s += __shfl_xor(s, 8);
        sc0[j] = s;
        float t = sc1[j];
        t += __shfl_xor(t, 1); t += __shfl_xor(t, 2);
        t += __shfl_xor(t, 4); t += __shfl_xor(t, 8);
        sc1[j] = t;
    }
    if (lr == 0) {
        int m0 = s0 + r0 + lh * 8;
        #pragma unroll
        for (int j = 0; j < 8; ++j)
            scores[b * S_LEN + m0 + j] = sc0[j] + vbias;
        int m1 = s0 + r0 + 16 + lh * 8;
        #pragma unroll
        for (int j = 0; j < 8; ++j)
            scores[b * S_LEN + m1 + j] = sc1[j] + vbias;
    }
}

// ---------------- K3a: per-batch max and 1/sum(exp) ----------------
__global__ __launch_bounds__(256) void stats_kernel(
    const float* __restrict__ scores, float* __restrict__ stats)
{
    __shared__ float red[256];
    int b = blockIdx.x, t = threadIdx.x;
    const float* sp = scores + b * S_LEN;
    float m = -1e30f;
    for (int i = t; i < S_LEN; i += 256) m = fmaxf(m, sp[i]);
    red[t] = m; __syncthreads();
    for (int o = 128; o > 0; o >>= 1) {
        if (t < o) red[t] = fmaxf(red[t], red[t + o]);
        __syncthreads();
    }
    m = red[0]; __syncthreads();
    float sum = 0.f;
    for (int i = t; i < S_LEN; i += 256) sum += __expf(sp[i] - m);
    red[t] = sum; __syncthreads();
    for (int o = 128; o > 0; o >>= 1) {
        if (t < o) red[t] += red[t + o];
        __syncthreads();
    }
    if (t == 0) { stats[b * 2] = m; stats[b * 2 + 1] = 1.0f / red[0]; }
}

// ---------------- K3b: weights + partial contexts (8 chunks x 32 batches) ----------------
__global__ __launch_bounds__(256) void ctx_partial_kernel(
    const float* __restrict__ enc, const float* __restrict__ scores,
    const float* __restrict__ stats, float* __restrict__ wout,
    float* __restrict__ part)
{
    __shared__ float wl[512];
    int c = blockIdx.x, b = blockIdx.y, t = threadIdx.x;
    const float m = stats[b * 2], inv = stats[b * 2 + 1];
    const float* sp = scores + b * S_LEN + c * 512;
    for (int i = t; i < 512; i += 256) {
        float w = __expf(sp[i] - m) * inv;
        wl[i] = w;
        wout[b * S_LEN + c * 512 + i] = w;
    }
    __syncthreads();
    const float* eb = enc + ((size_t)(b * S_LEN + c * 512)) * DIM;
    float acc = 0.f;
    for (int s = 0; s < 512; ++s) acc = fmaf(wl[s], eb[(size_t)s * DIM + t], acc);
    part[(b * 8 + c) * DIM + t] = acc;
}

// ---------------- K3c: reduce partial contexts ----------------
__global__ __launch_bounds__(256) void ctx_reduce_kernel(
    const float* __restrict__ part, float* __restrict__ ctx)
{
    int b = blockIdx.x, t = threadIdx.x;
    float acc = 0.f;
    #pragma unroll
    for (int c = 0; c < 8; ++c) acc += part[(b * 8 + c) * DIM + t];
    ctx[b * DIM + t] = acc;
}

extern "C" void kernel_launch(void* const* d_in, const int* in_sizes, int n_in,
                              void* d_out, int out_size, void* d_ws, size_t ws_size,
                              hipStream_t stream) {
    const float* dec = (const float*)d_in[0];
    const float* enc = (const float*)d_in[1];
    const float* w1  = (const float*)d_in[2];
    const float* b1  = (const float*)d_in[3];
    const float* w2  = (const float*)d_in[4];
    const float* b2  = (const float*)d_in[5];
    const float* vk  = (const float*)d_in[6];
    const float* vb  = (const float*)d_in[7];

    float* out_ctx = (float*)d_out;                 // [32*256]
    float* out_w   = out_ctx + NB * DIM;            // [32*4096]

    float* ws      = (float*)d_ws;
    float* qb      = ws;                            // 32*256
    float* scores  = qb + NB * DIM;                 // 32*4096
    float* stats   = scores + NB * S_LEN;           // 64
    float* part    = stats + 64;                    // 32*8*256
    unsigned short* w2bf = (unsigned short*)(part + NB * 8 * DIM); // 256*256 ushort

    qb_kernel<<<NB, 256, 0, stream>>>(dec, w1, b1, b2, qb);
    w2t_kernel<<<DIM, 256, 0, stream>>>(w2, w2bf);

    const size_t ldsBytes =
        (size_t)(ENC_ROWS + DIM) * LDS_STRIDE * sizeof(unsigned short) +
        2 * DIM * sizeof(float);
    score_kernel<<<dim3(S_LEN / ENC_ROWS, NB), 256, ldsBytes, stream>>>(
        enc, w2bf, qb, vk, vb, scores);

    stats_kernel<<<NB, 256, 0, stream>>>(scores, stats);
    ctx_partial_kernel<<<dim3(8, NB), 256, 0, stream>>>(enc, scores, stats, out_w, part);
    ctx_reduce_kernel<<<NB, 256, 0, stream>>>(part, out_ctx);
}
